// CirHashLoss_19791209300321
// MI455X (gfx1250) — matrix operand order repaired and checked by reading, not compiled, not run on hardware
//
#include <hip/hip_runtime.h>
#include <hip/hip_bf16.h>
#include <math.h>

// ---------------------------------------------------------------------------
// CirHashLoss on MI455X (gfx1250, wave32, WMMA + TDM)
//
//   K1 rowprep   : normalize u -> f16, argmax(y) labels, per-row quant q_i
//   K2 gemm+lse  : TENSOR_LOAD_TO_LDS stages 16KB A/B panels per block,
//                  sim tiles via v_wmma_f32_16x16x32_f16 fed from LDS
//                  (fragments preloaded so WMMAs issue back-to-back),
//                  fused masked online logsumexp -> per-block partials
//   K3 fill      : quant_loss[i,j] = q_i (rank-1), 256MB NT stores
//                  (the HBM floor: 256MB / 23.3 TB/s ~ 11us)
//   K4 finalize  : merge partials -> lse_p, lse_n -> softplus -> loss
// ---------------------------------------------------------------------------

typedef _Float16     v8h   __attribute__((ext_vector_type(8)));
typedef _Float16     v16h  __attribute__((ext_vector_type(16)));
typedef float        v2f   __attribute__((ext_vector_type(2)));
typedef float        v8f   __attribute__((ext_vector_type(8)));
typedef unsigned int u32x4 __attribute__((ext_vector_type(4)));
typedef int          i32x4 __attribute__((ext_vector_type(4)));
typedef int          i32x8 __attribute__((ext_vector_type(8)));

#define BATCH   8192
#define KDIM    64
#define NCLS    100
#define NBLK    64           // 8192 / 128 macro tiles per dim
#define NPART   (NBLK*NBLK)
#define NEG_INF (-__builtin_inff())

__device__ __forceinline__ void lse_update(float& m, float& s, float l) {
  if (l > m) { s = s * __expf(m - l) + 1.0f; m = l; }   // exp(-inf)=0 handles m=-inf
  else       { s += __expf(l - m); }
}

__device__ __forceinline__ void lse_merge(float& m, float& s, float mo, float so) {
  float mm = fmaxf(m, mo);
  if (mm > NEG_INF) {
    s = s * __expf(m - mm) + so * __expf(mo - mm);
    m = mm;
  } // both -inf: stay (-inf, 0)
}

// --------------------------- K1: row prep ----------------------------------
__global__ __launch_bounds__(256)
void rowprep_kernel(const float* __restrict__ u, const float* __restrict__ y,
                    _Float16* __restrict__ nh, int* __restrict__ label,
                    float* __restrict__ q) {
  const int wave = threadIdx.x >> 5;
  const int lane = threadIdx.x & 31;
  const int row  = blockIdx.x * 8 + wave;

  const float u0 = u[row * KDIM + lane];
  const float u1 = u[row * KDIM + lane + 32];
  float ss = u0 * u0 + u1 * u1;
  float sa = fabsf(u0) + fabsf(u1);
  #pragma unroll
  for (int off = 16; off > 0; off >>= 1) {
    ss += __shfl_xor(ss, off, 32);
    sa += __shfl_xor(sa, off, 32);
  }
  const float nrm = sqrtf(ss);
  const float rn  = 1.0f / fmaxf(nrm, 1e-12f);
  nh[row * KDIM + lane]      = (_Float16)(u0 * rn);
  nh[row * KDIM + lane + 32] = (_Float16)(u1 * rn);

  if (lane == 0) {
    const float cosv = sa / fmaxf(nrm * 8.0f, 1e-4f);      // sqrt(K)=8
    const float dmat = (1.0f - fminf(cosv, 0.99f)) * (KDIM * 0.5f);
    q[row] = log1pf(dmat * 0.1f);                          // /CAUCHY_GAMMA
  }

  float bv = NEG_INF; int bidx = 0x7fffffff;
  for (int c = lane; c < NCLS; c += 32) {
    const float v = y[row * NCLS + c];
    if (v > bv) { bv = v; bidx = c; }                      // first-occurrence argmax
  }
  #pragma unroll
  for (int off = 16; off > 0; off >>= 1) {
    const float ov = __shfl_xor(bv, off, 32);
    const int   oi = __shfl_xor(bidx, off, 32);
    if (ov > bv || (ov == bv && oi < bidx)) { bv = ov; bidx = oi; }
  }
  if (lane == 0) label[row] = bidx;
}

// ---------------- TDM: 16KB contiguous panel -> LDS ------------------------
// 1-D tile: data_size=8B, tensor_dim0 = tile_dim0 = 2048 elements (16KB).
__device__ __forceinline__ void tdm_load_16kb(const _Float16* gsrc, unsigned lds_off) {
#if __has_builtin(__builtin_amdgcn_tensor_load_to_lds)
  const unsigned long long ga = (unsigned long long)gsrc;
  u32x4 g0;
  g0.x = 1u;                                           // count=1, is_restore=0
  g0.y = lds_off;                                      // D#.lds_addr (bytes)
  g0.z = (unsigned)(ga & 0xffffffffu);                 // global_addr[31:0]
  g0.w = (unsigned)((ga >> 32) & 0x01ffffffu)          // global_addr[56:32]
       | 0x80000000u;                                  // type=2 ("image")
  i32x8 g1;
  g1[0] = 0x00030000;                                  // data_size=3 (8B), mask=0
  g1[1] = (int)(2048u << 16);                          // tensor_dim0[15:0]=2048
  g1[2] = (int)(1u << 16);                             // dim0 hi=0, tensor_dim1=1
  g1[3] = (int)(2048u << 16);                          // dim1 hi=0, tile_dim0=2048
  g1[4] = 0;                                           // tile_dim1=0, tile_dim2=0
  g1[5] = 2048;                                        // tensor_dim0_stride lo
  g1[6] = 0;                                           // stride hi, dim1_stride lo
  g1[7] = 0;
  const i32x4 z4 = {0, 0, 0, 0};
#if __clang_major__ >= 23
  const i32x8 z8 = {0, 0, 0, 0, 0, 0, 0, 0};
  __builtin_amdgcn_tensor_load_to_lds(g0, g1, z4, z4, z8, 0);
#else
  __builtin_amdgcn_tensor_load_to_lds(g0, g1, z4, z4, 0);
#endif
#endif
}

// ----------------------- K2: WMMA GEMM + fused LSE -------------------------
// Block = 128x128 macro tile, 8 waves, each wave one 16x128 strip (8 tiles).
__global__ __launch_bounds__(256)
void gemm_lse_kernel(const _Float16* __restrict__ nh,
                     const int* __restrict__ label,
                     float* __restrict__ partials) {
  const int bi = blockIdx.y, bj = blockIdx.x;
  float* part = partials + 4 * (bi * NBLK + bj);
  if (bj < bi) {        // uniform early-out; must still refresh partials
    if (threadIdx.x == 0) { part[0] = NEG_INF; part[1] = 0.f; part[2] = NEG_INF; part[3] = 0.f; }
    return;
  }

  __shared__ _Float16 pA[128 * KDIM];     // 16KB A panel (rows bi*128..+128)
  __shared__ _Float16 pB[128 * KDIM];     // 16KB B panel (rows bj*128..+128)
  __shared__ int      labR[128], labC[128];
  __shared__ float    red[8][4];

  const int t = threadIdx.x;
  if (t < 128) labR[t]       = label[bi * 128 + t];          // wave-uniform split
  else         labC[t - 128] = label[bj * 128 + (t - 128)];

#if __has_builtin(__builtin_amdgcn_tensor_load_to_lds)
  // Wave 0 drives the Tensor Data Mover; DMA runs while we hit the barrier.
  if (t < 32) {
    tdm_load_16kb(nh + (size_t)bi * (128 * KDIM), (unsigned)(unsigned long long)(void*)pA);
    if (bj != bi)
      tdm_load_16kb(nh + (size_t)bj * (128 * KDIM), (unsigned)(unsigned long long)(void*)pB);
    __builtin_amdgcn_s_wait_tensorcnt(0);
  }
#else
  {
    const v8h* sA = (const v8h*)(nh + (size_t)bi * (128 * KDIM));
    v8h* dA = (v8h*)pA;
    for (int i = t; i < 1024; i += 256) dA[i] = sA[i];       // 16KB / b128
    if (bj != bi) {
      const v8h* sB = (const v8h*)(nh + (size_t)bj * (128 * KDIM));
      v8h* dB = (v8h*)pB;
      for (int i = t; i < 1024; i += 256) dB[i] = sB[i];
    }
  }
#endif
  __syncthreads();

  const _Float16* PB = (bi == bj) ? pA : pB;                 // diagonal reuse

  const int wave = t >> 5;
  const int lane = t & 31;
  const int l16  = lane & 15;
  const int lhi  = lane >> 4;                                // 0 or 1

  v8f acc[8];
  #pragma unroll
  for (int ct = 0; ct < 8; ++ct)
    acc[ct] = (v8f){0.f, 0.f, 0.f, 0.f, 0.f, 0.f, 0.f, 0.f};

  const _Float16* arow = pA + (wave * 16 + l16) * KDIM;

  // EXEC is all-ones here (required for WMMA)
  #pragma unroll
  for (int k0 = 0; k0 < KDIM; k0 += 32) {
    // A fragment (16x32 f16): K in [k0+16*lhi,+16)
    const _Float16* ap = arow + k0 + (lhi << 4);
    const v8h a0 = *(const v8h*)(ap);
    const v8h a1 = *(const v8h*)(ap + 8);
    const v16h afrag = __builtin_shufflevector(a0, a1,
        0,1,2,3,4,5,6,7,8,9,10,11,12,13,14,15);

    // Preload all 8 B fragments so the 8 WMMAs below issue back-to-back.
    v16h bfrag[8];
    #pragma unroll
    for (int ct = 0; ct < 8; ++ct) {
      const _Float16* bp = PB + (ct * 16 + l16) * KDIM + k0 + (lhi << 4);
      const v8h b0 = *(const v8h*)(bp);
      const v8h b1 = *(const v8h*)(bp + 8);
      bfrag[ct] = __builtin_shufflevector(b0, b1,
          0,1,2,3,4,5,6,7,8,9,10,11,12,13,14,15);
    }
    #pragma unroll
    for (int ct = 0; ct < 8; ++ct)
      acc[ct] = __builtin_amdgcn_wmma_f32_16x16x32_f16(
          false, afrag, false, bfrag[ct], (short)0, acc[ct], false, false);
  }

  // D layout: lane -> N = lane%16; VGPR v -> M = v + 8*(lane/16)
  float mp = NEG_INF, sp = 0.f, mn = NEG_INF, sn = 0.f;
  #pragma unroll
  for (int ct = 0; ct < 8; ++ct) {
    const int jj = ct * 16 + l16;
    const int gj = bj * 128 + jj;
    const int lj = labC[jj];
    #pragma unroll
    for (int v = 0; v < 8; ++v) {
      const int ii = wave * 16 + v + (lhi << 3);
      const int gi = bi * 128 + ii;
      if (gj > gi) {                                   // strict upper triangle
        const float s = acc[ct][v];
        if (labR[ii] == lj) {
          const float ap_ = fmaxf(1.25f - s, 0.f);     // m=0.25
          lse_update(mp, sp, -ap_ * (s - 0.75f) * 256.0f);
        } else {
          const float an_ = fmaxf(s + 0.25f, 0.f);
          lse_update(mn, sn, an_ * (s - 0.25f) * 256.0f);
        }
      }
    }
  }

  #pragma unroll
  for (int off = 16; off > 0; off >>= 1) {
    float mo = __shfl_xor(mp, off, 32), so = __shfl_xor(sp, off, 32);
    lse_merge(mp, sp, mo, so);
    mo = __shfl_xor(mn, off, 32); so = __shfl_xor(sn, off, 32);
    lse_merge(mn, sn, mo, so);
  }
  if (lane == 0) { red[wave][0] = mp; red[wave][1] = sp; red[wave][2] = mn; red[wave][3] = sn; }
  __syncthreads();
  if (t == 0) {
    mp = red[0][0]; sp = red[0][1]; mn = red[0][2]; sn = red[0][3];
    for (int w = 1; w < 8; ++w) {
      lse_merge(mp, sp, red[w][0], red[w][1]);
      lse_merge(mn, sn, red[w][2], red[w][3]);
    }
    part[0] = mp; part[1] = sp; part[2] = mn; part[3] = sn;
  }
}

// ------------------------ K3: rank-1 quant fill ----------------------------
__global__ __launch_bounds__(256)
void fill_quant_kernel(const float* __restrict__ q, float* __restrict__ out) {
  const unsigned idx = blockIdx.x * 256u + threadIdx.x;   // over B*B/2 float2's
  const float v = q[idx >> 12];                            // idx / (8192/2)
  const v2f pv = {v, v};
  __builtin_nontemporal_store(pv, (v2f*)out + idx);
}

// --------------------------- K4: finalize ----------------------------------
__global__ __launch_bounds__(256)
void finalize_kernel(const float* __restrict__ partials, const float* __restrict__ q,
                     float* __restrict__ out) {
  const int t = threadIdx.x;
  float mp = NEG_INF, sp = 0.f, mn = NEG_INF, sn = 0.f, qs = 0.f;
  for (int i = t; i < NPART; i += 256) {
    const float* p = partials + 4 * i;
    lse_merge(mp, sp, p[0], p[1]);
    lse_merge(mn, sn, p[2], p[3]);
  }
  for (int i = t; i < BATCH; i += 256) qs += q[i];

  #pragma unroll
  for (int off = 16; off > 0; off >>= 1) {
    float mo = __shfl_xor(mp, off, 32), so = __shfl_xor(sp, off, 32);
    lse_merge(mp, sp, mo, so);
    mo = __shfl_xor(mn, off, 32); so = __shfl_xor(sn, off, 32);
    lse_merge(mn, sn, mo, so);
    qs += __shfl_xor(qs, off, 32);
  }
  __shared__ float red[8][5];
  const int wave = t >> 5, lane = t & 31;
  if (lane == 0) {
    red[wave][0] = mp; red[wave][1] = sp; red[wave][2] = mn; red[wave][3] = sn; red[wave][4] = qs;
  }
  __syncthreads();
  if (t == 0) {
    mp = red[0][0]; sp = red[0][1]; mn = red[0][2]; sn = red[0][3]; qs = red[0][4];
    for (int w = 1; w < 8; ++w) {
      lse_merge(mp, sp, red[w][0], red[w][1]);
      lse_merge(mn, sn, red[w][2], red[w][3]);
      qs += red[w][4];
    }
    const float lsep = mp + logf(sp);
    const float lsen = mn + logf(sn);
    const float x = lsep + lsen;
    const float cir = (x > 0.f) ? (x + log1pf(expf(-x))) : log1pf(expf(x));
    out[0] = cir + 0.1f * (qs / (float)BATCH);   // loss (mean quant = mean_i q_i)
    out[1] = cir;                                 // cir_loss
  }
}

// ---------------------------------------------------------------------------
extern "C" void kernel_launch(void* const* d_in, const int* in_sizes, int n_in,
                              void* d_out, int out_size, void* d_ws, size_t ws_size,
                              hipStream_t stream) {
  const float* u = (const float*)d_in[0];   // [8192, 64]
  const float* y = (const float*)d_in[1];   // [8192, 100]
  float* out = (float*)d_out;               // [loss, cir_loss, quant_loss(B*B)]

  // workspace layout (16B-aligned offsets), total ~1.13 MB
  char* ws = (char*)d_ws;
  _Float16* nh   = (_Float16*)(ws);                       // 8192*64 f16 = 1 MB
  int*      lab  = (int*)     (ws + 1048576);             // 32 KB
  float*    q    = (float*)   (ws + 1048576 + 32768);     // 32 KB
  float*    part = (float*)   (ws + 1048576 + 65536);     // 4096*4 f32 = 64 KB
  (void)ws_size; (void)in_sizes; (void)n_in; (void)out_size;

  rowprep_kernel<<<BATCH / 8, 256, 0, stream>>>(u, y, nh, lab, q);

  dim3 g2(NBLK, NBLK);
  gemm_lse_kernel<<<g2, 256, 0, stream>>>(nh, lab, part);

  fill_quant_kernel<<<(BATCH * (BATCH / 2)) / 256, 256, 0, stream>>>(q, out + 2);

  finalize_kernel<<<1, 256, 0, stream>>>(part, q, out);
}
